// StarTransformerClassifier_59382217835115
// MI455X (gfx1250) — matrix operand order repaired and checked by reading, not compiled, not run on hardware
//
#include <hip/hip_runtime.h>
#include <cmath>

// ---------------------------------------------------------------------------
// Types for CDNA5 WMMA (gfx1250, wave32)
// ---------------------------------------------------------------------------
typedef __attribute__((ext_vector_type(16))) __bf16 v16bf;
typedef __attribute__((ext_vector_type(8)))  float  v8f;

// Stage GEMM tiles with GLOBAL_LOAD_ASYNC_TO_LDS_B128 (ASYNCcnt) instead of
// global_load_b128 -> ds_store_b128 (LOADcnt + VGPR round trip).
#define USE_ASYNC_STAGING 1

__device__ __forceinline__ unsigned short f2bf(float f) {
  union { float f; unsigned u; } x; x.f = f;
  unsigned r = x.u + 0x7FFFu + ((x.u >> 16) & 1u);   // round-to-nearest-even
  return (unsigned short)(r >> 16);
}

// ---------------------------------------------------------------------------
// Elementwise f32 -> bf16 conversion (grid-stride)
// ---------------------------------------------------------------------------
__global__ __launch_bounds__(256)
void cvt_f32_bf16(const float* __restrict__ in, unsigned short* __restrict__ out,
                  long long n) {
  long long i  = (long long)blockIdx.x * blockDim.x + threadIdx.x;
  long long st = (long long)gridDim.x * blockDim.x;
  for (; i < n; i += st) out[i] = f2bf(in[i]);
}

// ---------------------------------------------------------------------------
// Transposing f32 -> bf16 conversion: in (Z,R,C) row-major -> out (Z,C,R)
// 32x32 LDS tiles, block 256 = 32x8. R, C must be multiples of 32.
// ---------------------------------------------------------------------------
__global__ __launch_bounds__(256)
void cvt_t_f32_bf16(const float* __restrict__ in, unsigned short* __restrict__ out,
                    int R, int C, long long sIn, long long sOut) {
  __shared__ unsigned short t[32][33];
  const int z = blockIdx.z;
  in  += (size_t)z * sIn;
  out += (size_t)z * sOut;
  const int r0 = blockIdx.y * 32, c0 = blockIdx.x * 32;
  const int tx = threadIdx.x & 31, ty = threadIdx.x >> 5;
#pragma unroll
  for (int i = 0; i < 4; ++i)
    t[ty + 8 * i][tx] = f2bf(in[(size_t)(r0 + ty + 8 * i) * C + c0 + tx]);
  __syncthreads();
#pragma unroll
  for (int i = 0; i < 4; ++i)
    out[(size_t)(c0 + ty + 8 * i) * R + r0 + tx] = t[tx][ty + 8 * i];
}

// ---------------------------------------------------------------------------
// bf16 WMMA GEMM:  C[M,N] = A[M,K] @ B[K,N] (+bias) (+res) (relu)
//   - A row-major bf16 (M x K, ld = lda)
//   - Bt = B TRANSPOSED, row-major bf16 (N x K, ld = ldb)  [pre-transposed]
//   - block 256 (8 wave32s), C tile 128x128, wave tile 32x64, K step 32
//   - double-buffered LDS; async tile staging direct to LDS (TENSOR-free
//     CDNA5 async path, ASYNCcnt); padded LDS K-stride 40 (80B rows)
// Requires: M % 128 == 0, N % 128 == 0, K % 32 == 0.
// ---------------------------------------------------------------------------
#define GTM 128
#define GTN 128
#define GTK 32
#define GKP 40   // padded LDS K-stride in bf16 elements (80B rows)

template <bool BIAS, bool RES, bool RELU, bool OF32, bool OBF16>
__global__ __launch_bounds__(256)
void gemm_wmma_bf16(const unsigned short* __restrict__ A, int lda, long long sA,
                    const unsigned short* __restrict__ Bt, int ldb, long long sB,
                    const float* __restrict__ bias,
                    const float* __restrict__ res, int ldr, long long sR,
                    float* __restrict__ Cf, unsigned short* __restrict__ Cb,
                    int ldc, long long sC, int K) {
  __shared__ unsigned short As[2][GTM * GKP];
  __shared__ unsigned short Bs[2][GTN * GKP];

  const int z = blockIdx.z;
  A  += (size_t)z * sA;
  Bt += (size_t)z * sB;
  if constexpr (RES)   res += (size_t)z * sR;
  if constexpr (OF32)  Cf  += (size_t)z * sC;
  if constexpr (OBF16) Cb  += (size_t)z * sC;

  const int tid   = threadIdx.x;
  const int lane  = tid & 31;
  const int wave  = tid >> 5;
  const int lrow  = lane & 15;   // row within 16-lane half
  const int lhalf = lane >> 4;   // which half of the wave
  const int waveM = wave & 3;    // 4 waves along M
  const int waveN = wave >> 2;   // 2 waves along N
  const int rowBase = blockIdx.y * GTM;
  const int colBase = blockIdx.x * GTN;

  // staging indices: both tiles are 128 rows x 32 K (16 elems / thread)
  const int sr = tid >> 1;          // 0..127 (A row / Bt row)
  const int sc = (tid & 1) << 4;    // 0 or 16 (K offset)

  v8f acc[2][4];
#pragma unroll
  for (int i = 0; i < 2; ++i)
#pragma unroll
    for (int j = 0; j < 4; ++j)
#pragma unroll
      for (int r = 0; r < 8; ++r) acc[i][j][r] = 0.0f;

#if USE_ASYNC_STAGING
  // Memory -> LDS without the VGPR round trip. INST_OFFSET is added to both
  // the LDS and global addresses (ISA 15.18), so offset:16 stages the second
  // 16B half with the same address registers. Tracked by ASYNCcnt.
  auto stage = [&](int kt, int buf) {
    const unsigned short* ap = A + (size_t)(rowBase + sr) * lda + kt * GTK + sc;
    const unsigned short* bp = Bt + (size_t)(colBase + sr) * ldb + kt * GTK + sc;
    unsigned la = (unsigned)(unsigned long long)&As[buf][sr * GKP + sc];
    unsigned lb = (unsigned)(unsigned long long)&Bs[buf][sr * GKP + sc];
    unsigned long long ga = (unsigned long long)ap;
    unsigned long long gb = (unsigned long long)bp;
    asm volatile(
        "global_load_async_to_lds_b128 %0, %1, off\n\t"
        "global_load_async_to_lds_b128 %0, %1, off offset:16\n\t"
        "global_load_async_to_lds_b128 %2, %3, off\n\t"
        "global_load_async_to_lds_b128 %2, %3, off offset:16"
        :: "v"(la), "v"(ga), "v"(lb), "v"(gb)
        : "memory");
    // keep L2 one tile ahead of the double buffer
    __builtin_prefetch(ap + GTK, 0, 0);
    __builtin_prefetch(bp + GTK, 0, 0);
  };
  auto stage_wait = [&]() {
    asm volatile("s_wait_asynccnt 0x0" ::: "memory");
  };
#else
  uint4 pa0, pa1, pb0, pb1;
  auto gload = [&](int kt) {
    const unsigned short* ap = A + (size_t)(rowBase + sr) * lda + kt * GTK + sc;
    pa0 = ((const uint4*)ap)[0];
    pa1 = ((const uint4*)ap)[1];
    const unsigned short* bp = Bt + (size_t)(colBase + sr) * ldb + kt * GTK + sc;
    pb0 = ((const uint4*)bp)[0];
    pb1 = ((const uint4*)bp)[1];
    __builtin_prefetch(ap + GTK, 0, 0);
    __builtin_prefetch(bp + GTK, 0, 0);
  };
  auto sstore = [&](int buf) {
    uint4* as = (uint4*)&As[buf][sr * GKP + sc];
    as[0] = pa0; as[1] = pa1;
    uint4* bs = (uint4*)&Bs[buf][sr * GKP + sc];
    bs[0] = pb0; bs[1] = pb1;
  };
#endif

  union Frag { v16bf v; uint4 q[2]; };

  auto compute = [&](int buf) {
    Frag a[2], b[4];
    // A 16x32 fragment, ISA layout: lanes 0-15 K in [0,8)+[16,24),
    //                               lanes 16-31 K in [8,16)+[24,32)
#pragma unroll
    for (int mi = 0; mi < 2; ++mi) {
      const unsigned short* p =
          &As[buf][(waveM * 32 + mi * 16 + lrow) * GKP + lhalf * 8];
      a[mi].q[0] = *(const uint4*)p;
      a[mi].q[1] = *(const uint4*)(p + 16);
    }
    // B 32x16 fragment: lanes 0-15 K in [0,8)+[16,24), lanes 16-31 K in [8,16)+[24,32)
#pragma unroll
    for (int ni = 0; ni < 4; ++ni) {
      const unsigned short* p =
          &Bs[buf][(waveN * 64 + ni * 16 + lrow) * GKP + lhalf * 8];
      b[ni].q[0] = *(const uint4*)p;
      b[ni].q[1] = *(const uint4*)(p + 16);
    }
#pragma unroll
    for (int mi = 0; mi < 2; ++mi)
#pragma unroll
      for (int ni = 0; ni < 4; ++ni)
        acc[mi][ni] = __builtin_amdgcn_wmma_f32_16x16x32_bf16(
            false, a[mi].v, false, b[ni].v, (short)0, acc[mi][ni],
            false, false);
  };

  const int nk = K / GTK;
#if USE_ASYNC_STAGING
  stage(0, 0);
  stage_wait();
  __syncthreads();
#pragma unroll 1
  for (int kt = 0; kt < nk; ++kt) {
    const int cur = kt & 1;
    if (kt + 1 < nk) stage(kt + 1, cur ^ 1);  // async fill of other buffer
    compute(cur);                             // WMMAs overlap the DMA
    if (kt + 1 < nk) stage_wait();            // own async writes done
    __syncthreads();                          // cross-wave visibility
  }
#else
  gload(0);
  sstore(0);
  __syncthreads();
#pragma unroll 1
  for (int kt = 0; kt < nk; ++kt) {
    const int cur = kt & 1;
    if (kt + 1 < nk) gload(kt + 1);
    compute(cur);
    if (kt + 1 < nk) sstore(cur ^ 1);
    __syncthreads();
  }
#endif

  // Epilogue. C fragment layout: VGPR r -> M = r + 8*lhalf, N = lane&15
#pragma unroll
  for (int mi = 0; mi < 2; ++mi)
#pragma unroll
    for (int ni = 0; ni < 4; ++ni) {
      const int col = colBase + waveN * 64 + ni * 16 + lrow;
      float bv = 0.0f;
      if constexpr (BIAS) bv = bias[col];
#pragma unroll
      for (int r = 0; r < 8; ++r) {
        const int row = rowBase + waveM * 32 + mi * 16 + lhalf * 8 + r;
        float v = acc[mi][ni][r] + bv;
        if constexpr (RES)  v += res[(size_t)row * ldr + col];
        if constexpr (RELU) v = fmaxf(v, 0.0f);
        if constexpr (OF32)  Cf[(size_t)row * ldc + col] = v;
        if constexpr (OBF16) Cb[(size_t)row * ldc + col] = f2bf(v);
      }
    }
}

// ---------------------------------------------------------------------------
// Tiny single-head attention: seq len 4 over "batch" n in [0,512)
// scores[l][m] = q[l]·k[m] / sqrt(D); softmax over m; out = attn @ v -> bf16
// ---------------------------------------------------------------------------
__global__ __launch_bounds__(256)
void attn_small(const float* __restrict__ Q, int ldq,
                const float* __restrict__ Kp, int ldk,
                const float* __restrict__ Vp, int ldv,
                unsigned short* __restrict__ O, int ldo, float scale) {
  constexpr int S = 512, D = 1536;
  const int n = blockIdx.x, tid = threadIdx.x;
  __shared__ float red[16][256];
  __shared__ float aw[16];

  float s[4][4];
#pragma unroll
  for (int i = 0; i < 4; ++i)
#pragma unroll
    for (int j = 0; j < 4; ++j) s[i][j] = 0.0f;

  for (int d = tid; d < D; d += 256) {
    float qv[4], kv[4];
#pragma unroll
    for (int l = 0; l < 4; ++l) qv[l] = Q[(size_t)(l * S + n) * ldq + d];
#pragma unroll
    for (int m = 0; m < 4; ++m) kv[m] = Kp[(size_t)(m * S + n) * ldk + d];
#pragma unroll
    for (int l = 0; l < 4; ++l)
#pragma unroll
      for (int m = 0; m < 4; ++m) s[l][m] += qv[l] * kv[m];
  }
#pragma unroll
  for (int i = 0; i < 16; ++i) red[i][tid] = s[i >> 2][i & 3];
  __syncthreads();
  if (tid < 16) {
    float t = 0.0f;
    for (int j = 0; j < 256; ++j) t += red[tid][j];
    aw[tid] = t * scale;
  }
  __syncthreads();
  if (tid < 4) {
    const int l = tid;
    float mx = aw[l * 4];
#pragma unroll
    for (int m = 1; m < 4; ++m) mx = fmaxf(mx, aw[l * 4 + m]);
    float e[4], se = 0.0f;
#pragma unroll
    for (int m = 0; m < 4; ++m) { e[m] = __expf(aw[l * 4 + m] - mx); se += e[m]; }
#pragma unroll
    for (int m = 0; m < 4; ++m) aw[l * 4 + m] = e[m] / se;
  }
  __syncthreads();
  for (int d = tid; d < D; d += 256) {
    float vv[4];
#pragma unroll
    for (int m = 0; m < 4; ++m) vv[m] = Vp[(size_t)(m * S + n) * ldv + d];
#pragma unroll
    for (int l = 0; l < 4; ++l) {
      float o = aw[l * 4 + 0] * vv[0] + aw[l * 4 + 1] * vv[1] +
                aw[l * 4 + 2] * vv[2] + aw[l * 4 + 3] * vv[3];
      O[(size_t)(l * S + n) * ldo + d] = f2bf(o);
    }
  }
}

// ---------------------------------------------------------------------------
// Row LayerNorm (D per row), writes f32 and/or bf16. One block per row.
// ---------------------------------------------------------------------------
__global__ __launch_bounds__(256)
void ln_rows(const float* __restrict__ in, const float* __restrict__ g,
             const float* __restrict__ b, float* __restrict__ outf,
             unsigned short* __restrict__ outb, int D) {
  const int row = blockIdx.x, tid = threadIdx.x;
  const float* x = in + (size_t)row * D;
  float s = 0.0f, s2 = 0.0f;
  for (int d = tid; d < D; d += 256) { float v = x[d]; s += v; s2 += v * v; }
  __shared__ float rs[256], rs2[256];
  rs[tid] = s; rs2[tid] = s2;
  __syncthreads();
  for (int off = 128; off > 0; off >>= 1) {
    if (tid < off) { rs[tid] += rs[tid + off]; rs2[tid] += rs2[tid + off]; }
    __syncthreads();
  }
  const float mean = rs[0] / D;
  const float var  = rs2[0] / D - mean * mean;
  const float inv  = rsqrtf(var + 1e-5f);
  for (int d = tid; d < D; d += 256) {
    float v = (x[d] - mean) * inv * g[d] + b[d];
    if (outf) outf[(size_t)row * D + d] = v;
    if (outb) outb[(size_t)row * D + d] = f2bf(v);
  }
}

// ---------------------------------------------------------------------------
// Pairwise center scores: out[b][p] = c[b, fi[p]] . ec[b, si[p]]
// triu_indices(NC,1) row-major pair ordering. One block per (p, b).
// ---------------------------------------------------------------------------
__global__ __launch_bounds__(128)
void pair_scores(const float* __restrict__ c, const float* __restrict__ ec,
                 float* __restrict__ out, int NC, int D, int NP) {
  const int p = blockIdx.x, b = blockIdx.y;
  int i = 0, rem = p;
  while (rem >= NC - 1 - i) { rem -= NC - 1 - i; ++i; }
  const int j = i + 1 + rem;
  const float* ci = c  + ((size_t)b * NC + i) * D;
  const float* ej = ec + ((size_t)b * NC + j) * D;
  float s = 0.0f;
  for (int d = threadIdx.x; d < D; d += 128) s += ci[d] * ej[d];
  __shared__ float r[128];
  r[threadIdx.x] = s;
  __syncthreads();
  for (int off = 64; off > 0; off >>= 1) {
    if (threadIdx.x < off) r[threadIdx.x] += r[threadIdx.x + off];
    __syncthreads();
  }
  if (threadIdx.x == 0) out[(size_t)b * NP + p] = r[0];
}

// ---------------------------------------------------------------------------
// Host orchestration
// ---------------------------------------------------------------------------
extern "C" void kernel_launch(void* const* d_in, const int* in_sizes, int n_in,
                              void* d_out, int out_size, void* d_ws,
                              size_t ws_size, hipStream_t stream) {
  (void)in_sizes; (void)n_in; (void)out_size; (void)ws_size;

  constexpr long long NB = 4, S = 512, D = 1536, FF = 2048, NC = 128;
  constexpr long long MT = NB * S;                 // 2048 token rows
  constexpr long long NP = NC * (NC - 1) / 2;      // 8128 pairs
  constexpr int D3 = 3 * (int)D;                   // 4608

  const float* x   = (const float*)d_in[0];
  const float* cm  = (const float*)d_in[1];
  // d_in[2] neighbor_map: unused by the reference
  const float* wsi = (const float*)d_in[3];
  const float* bsi = (const float*)d_in[4];
  const float* wso = (const float*)d_in[5];
  const float* bso = (const float*)d_in[6];
  const float* wci = (const float*)d_in[7];
  const float* bci = (const float*)d_in[8];
  const float* wco = (const float*)d_in[9];
  const float* bco = (const float*)d_in[10];
  const float* g1  = (const float*)d_in[11];
  const float* be1 = (const float*)d_in[12];
  const float* g2  = (const float*)d_in[13];
  const float* be2 = (const float*)d_in[14];
  const float* g3  = (const float*)d_in[15];
  const float* be3 = (const float*)d_in[16];
  const float* wf1 = (const float*)d_in[17];
  const float* bf1 = (const float*)d_in[18];
  const float* wf2 = (const float*)d_in[19];
  const float* bf2 = (const float*)d_in[20];

  // ---- workspace bump allocator (~142 MB total) ----
  char* ws = (char*)d_ws;
  size_t off = 0;
  auto alloc = [&](size_t bytes) -> char* {
    char* p = ws + off;
    off = (off + bytes + 255) & ~(size_t)255;
    return p;
  };
  unsigned short* xb   = (unsigned short*)alloc(MT * D * 2);     // x bf16 (A)
  unsigned short* xbT  = (unsigned short*)alloc(NB * D * S * 2); // x^T bf16 (B)
  unsigned short* wsiT = (unsigned short*)alloc((size_t)D3 * D * 2);
  unsigned short* wsoT = (unsigned short*)alloc(D * D * 2);
  unsigned short* wciT = (unsigned short*)alloc((size_t)D3 * D * 2);
  unsigned short* wcoT = (unsigned short*)alloc(D * D * 2);
  unsigned short* wf1T = (unsigned short*)alloc(FF * D * 2);     // (2048,1536)
  unsigned short* wf2T = (unsigned short*)alloc(D * FF * 2);     // (1536,2048)
  unsigned short* cmB  = (unsigned short*)alloc(NB * NC * S * 2);
  unsigned short* actB = (unsigned short*)alloc(MT * FF * 2);    // attn/ffn act
  unsigned short* hB   = (unsigned short*)alloc(MT * D * 2);     // h bf16; embs^T
  float* qkvF = (float*)alloc(MT * D3 * 4);                      // qkv / (qc,kvc)
  float* preF = (float*)alloc(MT * D * 4);                       // pre-LN buffer
  float* hF   = (float*)alloc(MT * D * 4);                       // h f32 / embs f32
  float* cF   = (float*)alloc(NB * NC * D * 4);
  float* ecF  = (float*)alloc(NB * NC * D * 4);

  auto cvt = [&](const float* src, unsigned short* dst, long long n) {
    cvt_f32_bf16<<<2048, 256, 0, stream>>>(src, dst, n);
  };
  auto cvtT = [&](const float* src, unsigned short* dst, int R, int C, int Z,
                  long long sIn, long long sOut) {
    dim3 g((unsigned)(C / 32), (unsigned)(R / 32), (unsigned)Z);
    cvt_t_f32_bf16<<<g, 256, 0, stream>>>(src, dst, R, C, sIn, sOut);
  };
  auto grid = [&](int M, int N, int Z) {
    return dim3((unsigned)(N / 128), (unsigned)(M / 128), (unsigned)Z);
  };

  const float scale = 1.0f / sqrtf((float)D);

  // 0) bf16 conversions; all GEMM B operands pre-transposed to N-major
  cvt(x, xb, MT * D);
  cvt(cm, cmB, NB * NC * S);
  cvtT(x,   xbT,  (int)S, (int)D, (int)NB, S * D, D * S);
  cvtT(wsi, wsiT, (int)D, D3, 1, 0, 0);
  cvtT(wso, wsoT, (int)D, (int)D, 1, 0, 0);
  cvtT(wci, wciT, (int)D, D3, 1, 0, 0);
  cvtT(wco, wcoT, (int)D, (int)D, 1, 0, 0);
  cvtT(wf1, wf1T, (int)D, (int)FF, 1, 0, 0);
  cvtT(wf2, wf2T, (int)FF, (int)D, 1, 0, 0);

  // 1) self-attn qkv = x @ w_self_in + b -> f32 (2048 x 4608)
  gemm_wmma_bf16<true, false, false, true, false>
      <<<grid((int)MT, D3, 1), 256, 0, stream>>>(
          xb, (int)D, 0, wsiT, (int)D, 0, bsi, nullptr, 0, 0,
          qkvF, nullptr, D3, 0, (int)D);

  // 2) 4x4 attention + softmax -> bf16 attn-out
  attn_small<<<(unsigned)S, 256, 0, stream>>>(
      qkvF, D3, qkvF + D, D3, qkvF + 2 * D, D3, actB, (int)D, scale);

  // 3) out-proj + residual(x) -> preF; LN1 -> hF (f32) + hB (bf16)
  gemm_wmma_bf16<true, true, false, true, false>
      <<<grid((int)MT, (int)D, 1), 256, 0, stream>>>(
          actB, (int)D, 0, wsoT, (int)D, 0, bso, x, (int)D, 0,
          preF, nullptr, (int)D, 0, (int)D);
  ln_rows<<<(unsigned)MT, 256, 0, stream>>>(preF, g1, be1, hF, hB, (int)D);

  // 4) cross-attn: q = h1 @ wq_cross ; k,v = x @ w_cross_in[:,1536:]
  float* qcF  = qkvF;              // 2048 x 1536
  float* kvcF = qkvF + MT * D;     // 2048 x 3072
  gemm_wmma_bf16<true, false, false, true, false>
      <<<grid((int)MT, (int)D, 1), 256, 0, stream>>>(
          hB, (int)D, 0, wciT, (int)D, 0, bci, nullptr, 0, 0,
          qcF, nullptr, (int)D, 0, (int)D);
  gemm_wmma_bf16<true, false, false, true, false>
      <<<grid((int)MT, 2 * (int)D, 1), 256, 0, stream>>>(
          xb, (int)D, 0, wciT + (size_t)D * D, (int)D, 0, bci + D,
          nullptr, 0, 0, kvcF, nullptr, 2 * (int)D, 0, (int)D);
  attn_small<<<(unsigned)S, 256, 0, stream>>>(
      qcF, (int)D, kvcF, 2 * (int)D, kvcF + D, 2 * (int)D, actB, (int)D, scale);

  // 5) cross out-proj + residual(h1) -> preF; LN2 -> hF + hB
  gemm_wmma_bf16<true, true, false, true, false>
      <<<grid((int)MT, (int)D, 1), 256, 0, stream>>>(
          actB, (int)D, 0, wcoT, (int)D, 0, bco, hF, (int)D, 0,
          preF, nullptr, (int)D, 0, (int)D);
  ln_rows<<<(unsigned)MT, 256, 0, stream>>>(preF, g2, be2, hF, hB, (int)D);

  // 6) FFN: relu(h2 @ w_ff1 + b) stored as bf16, then @ w_ff2 + res(h2)
  gemm_wmma_bf16<true, false, true, false, true>
      <<<grid((int)MT, (int)FF, 1), 256, 0, stream>>>(
          hB, (int)D, 0, wf1T, (int)D, 0, bf1, nullptr, 0, 0,
          nullptr, actB, (int)FF, 0, (int)D);
  gemm_wmma_bf16<true, true, false, true, false>
      <<<grid((int)MT, (int)D, 1), 256, 0, stream>>>(
          actB, (int)FF, 0, wf2T, (int)FF, 0, bf2, hF, (int)D, 0,
          preF, nullptr, (int)D, 0, (int)FF);
  // LN3 -> embs f32 into hF, then transpose-convert to bf16 (B operand)
  ln_rows<<<(unsigned)MT, 256, 0, stream>>>(preF, g3, be3, hF, nullptr, (int)D);
  cvtT(hF, hB /* now embs^T (4,1536,512) */, (int)S, (int)D, (int)NB,
       S * D, D * S);

  // 7) center aggregation (batched): c = cm @ x ; ec = cm @ embs
  gemm_wmma_bf16<false, false, false, true, false>
      <<<grid((int)NC, (int)D, (int)NB), 256, 0, stream>>>(
          cmB, (int)S, NC * S, xbT, (int)S, D * S, nullptr, nullptr, 0, 0,
          cF, nullptr, (int)D, NC * D, (int)S);
  gemm_wmma_bf16<false, false, false, true, false>
      <<<grid((int)NC, (int)D, (int)NB), 256, 0, stream>>>(
          cmB, (int)S, NC * S, hB, (int)S, D * S, nullptr, nullptr, 0, 0,
          ecF, nullptr, (int)D, NC * D, (int)S);

  // 8) pairwise upper-triangle dot products -> output (4 x 8128 f32)
  pair_scores<<<dim3((unsigned)NP, (unsigned)NB), 128, 0, stream>>>(
      cF, ecF, (float*)d_out, (int)NC, (int)D, (int)NP);
}
